// T10_transformer_70849780514971
// MI455X (gfx1250) — matrix operand order repaired and checked by reading, not compiled, not run on hardware
//
#include <hip/hip_runtime.h>

#define BATCH 16384
#define TT 64
#define HH 64
#define REGD 300
#define KP2 320   // padded K for layer 2

typedef _Float16 v16h __attribute__((ext_vector_type(16)));
typedef _Float16 h8   __attribute__((ext_vector_type(8)));
typedef float    v8f  __attribute__((ext_vector_type(8)));

__device__ __forceinline__ float sigmoidf_(float x) { return 1.0f / (1.0f + __expf(-x)); }
__device__ __forceinline__ float eluf_(float x)     { return x > 0.0f ? x : (__expf(x) - 1.0f); }

__device__ __forceinline__ v16h zero_v16h() {
  v16h z = {};
  return z;
}

// A fragment (16x32 f16) from a row-major [rows][stride] f16 buffer in LDS.
// half i: K = kbase + 16*hiHalf + i, M = jl (lane&15)
__device__ __forceinline__ v16h ld_afrag_lds(const _Float16* base, int stride, int M, int hiHalf, int kbase) {
  const h8 lo = *(const h8*)(base + M * stride + kbase + 16 * hiHalf);
  const h8 hv = *(const h8*)(base + M * stride + kbase + 16 * hiHalf + 8);
  v16h a;
#pragma unroll
  for (int i = 0; i < 8; ++i) { a[i] = lo[i]; a[i + 8] = hv[i]; }
  return a;
}

// B fragment (32x16 f16) from a row-major f16 weight buffer in LDS:
// lanes 0-15 hold K=kbase..+15 (N=lane), lanes 16-31 hold K=kbase+16..+31.
// Caller passes kstart = kbase + 16*hiHalf; reads 16 contiguous halves of W[n][kstart..].
__device__ __forceinline__ v16h ld_bfrag_lds(const _Float16* W, int ld, int n, int kstart) {
  const h8 lo = *(const h8*)(W + n * ld + kstart);
  const h8 hv = *(const h8*)(W + n * ld + kstart + 8);
  v16h b;
#pragma unroll
  for (int i = 0; i < 8; ++i) { b[i] = lo[i]; b[i + 8] = hv[i]; }
  return b;
}

// A fragment from global f32 row (one batch row, 64 contiguous f32)
__device__ __forceinline__ v16h ld_afrag_gf32(const float* __restrict__ row, int hiHalf, int kbase) {
  const float4* p0 = (const float4*)(row + kbase + 16 * hiHalf);
  const float4* p1 = (const float4*)(row + kbase + 16 * hiHalf + 8);
  float4 q0 = p0[0], q1 = p0[1], q2 = p1[0], q3 = p1[1];
  v16h a;
  a[0]=(_Float16)q0.x; a[1]=(_Float16)q0.y; a[2]=(_Float16)q0.z; a[3]=(_Float16)q0.w;
  a[4]=(_Float16)q1.x; a[5]=(_Float16)q1.y; a[6]=(_Float16)q1.z; a[7]=(_Float16)q1.w;
  a[8]=(_Float16)q2.x; a[9]=(_Float16)q2.y; a[10]=(_Float16)q2.z; a[11]=(_Float16)q2.w;
  a[12]=(_Float16)q3.x; a[13]=(_Float16)q3.y; a[14]=(_Float16)q3.z; a[15]=(_Float16)q3.w;
  return a;
}

// B fragment built from global f32 weights (row-major [n][ld]); kstart = kbase + 16*hiHalf
__device__ __forceinline__ v16h ld_bfrag_g(const float* __restrict__ W, int ld, int n, int kstart) {
  const float4* p = (const float4*)(W + (size_t)n * ld + kstart);
  float4 q0 = p[0], q1 = p[1], q2 = p[2], q3 = p[3];
  v16h b;
  b[0]=(_Float16)q0.x; b[1]=(_Float16)q0.y; b[2]=(_Float16)q0.z; b[3]=(_Float16)q0.w;
  b[4]=(_Float16)q1.x; b[5]=(_Float16)q1.y; b[6]=(_Float16)q1.z; b[7]=(_Float16)q1.w;
  b[8]=(_Float16)q2.x; b[9]=(_Float16)q2.y; b[10]=(_Float16)q2.z; b[11]=(_Float16)q2.w;
  b[12]=(_Float16)q3.x; b[13]=(_Float16)q3.y; b[14]=(_Float16)q3.z; b[15]=(_Float16)q3.w;
  return b;
}

#define WMMA_F16(A, B, C) \
  __builtin_amdgcn_wmma_f32_16x16x32_f16(false, (A), false, (B), (short)0, (C), false, false)

// ---------------------------------------------------------------------------
// Phase 1: GRU-D scan + streaming softmax pooling.  1 wave = 16 batch rows.
// Recurrent weights live as f16 in LDS (shared by the 4 waves of the block);
// B fragments are pulled with ds_load_b128 right before each WMMA so their
// live ranges are short and nothing spills.
// ---------------------------------------------------------------------------
__global__ __launch_bounds__(128) void grud_scan_kernel(
    const float* __restrict__ Xfa, const float* __restrict__ Xlast,
    const float* __restrict__ Xmean, const float* __restrict__ Xmask,
    const float* __restrict__ Xdelta,
    const float* __restrict__ Wgx, const float* __restrict__ bgx,
    const float* __restrict__ Wgh, const float* __restrict__ bgh,
    const float* __restrict__ Wxc, const float* __restrict__ bxc,
    const float* __restrict__ Whn, const float* __restrict__ Whc,
    const float* __restrict__ Wmc,
    const float* __restrict__ WsT, const float* __restrict__ WsM,
    float* __restrict__ out_g,   // [B,T,H]
    float* __restrict__ xgru_g,  // [B,T]
    float* __restrict__ hidT_g,  // [B,H] (workspace)
    float* __restrict__ hidM_g)  // [B,H] (workspace)
{
  __shared__ _Float16 sWhc[128 * HH];   // 16 KB, f16 copy of W_hc
  __shared__ _Float16 sWhn[HH * HH];    // 8 KB,  f16 copy of W_hn
  __shared__ _Float16 sH[4][16 * HH];   // 8 KB
  __shared__ _Float16 sR[4][16 * HH];   // 8 KB
  __shared__ float sD[4][16];
  __shared__ float sX[4][16];
  __shared__ float sM[4][16];

  // cooperative staging of recurrent weights into LDS (f16)
  for (int i = threadIdx.x; i < 128 * HH; i += 128) sWhc[i] = (_Float16)Whc[i];
  for (int i = threadIdx.x; i < HH * HH;  i += 128) sWhn[i] = (_Float16)Whn[i];
  __syncthreads();

  const int lane = threadIdx.x & 31;
  const int wv   = threadIdx.x >> 5;
  const int jl   = lane & 15;
  const int hiH  = lane >> 4;  // 0 or 1
  const int rowBase = (blockIdx.x * 4 + wv) * 16;

  _Float16* hA = sH[wv];
  _Float16* rA = sR[wv];
  float* scD = sD[wv];
  float* scX = sX[wv];
  float* scM = sM[wv];

  // per-lane column constants (N = ct*16 + jl)
  float ghw[4], ghb[4];
  float xw0[4], xw1[4], xw2[4], xb0[4], xb1[4], xb2[4];
  float mw0[4], mw1[4], mw2[4], stw[4], smw[4];
#pragma unroll
  for (int ct = 0; ct < 4; ++ct) {
    int j = ct * 16 + jl;
    ghw[ct] = Wgh[j];        ghb[ct] = bgh[j];
    xw0[ct] = Wxc[j];        xw1[ct] = Wxc[HH + j];     xw2[ct] = Wxc[2 * HH + j];
    xb0[ct] = bxc[j];        xb1[ct] = bxc[HH + j];     xb2[ct] = bxc[2 * HH + j];
    mw0[ct] = Wmc[j];        mw1[ct] = Wmc[HH + j];     mw2[ct] = Wmc[2 * HH + j];
    stw[ct] = WsT[j];        smw[ct] = WsM[j];
  }
  const float wgx = Wgx[0], bg0 = bgx[0];

  // h state + streaming-softmax accumulators in WMMA C layout
  float hC[4][8], accT[4][8], accM[4][8], ZT[8], ZM[8];
#pragma unroll
  for (int ct = 0; ct < 4; ++ct)
#pragma unroll
    for (int v = 0; v < 8; ++v) { hC[ct][v] = 0.f; accT[ct][v] = 0.f; accM[ct][v] = 0.f; }
#pragma unroll
  for (int v = 0; v < 8; ++v) { ZT[v] = 0.f; ZM[v] = 0.f; }

#pragma unroll 1
  for (int t = 0; t < TT; ++t) {
    // ---- per-row scalars (lanes 0..15 = rows) ----
    if (lane < 16) {
      const int row = rowBase + lane;
      const int idx = row * TT + t;
      float d  = Xdelta[idx];
      float m  = Xmask[idx];
      float xi = Xfa[idx];
      float xl = Xlast[idx];
      float xm = Xmean[idx];
      float gx = __expf(-fmaxf(d * wgx + bg0, 0.f));
      float X  = m * xi + (1.f - m) * (gx * xl + (1.f - gx) * xm);
      scD[lane] = d; scX[lane] = X; scM[lane] = m;
      xgru_g[idx] = X;
    }
    __builtin_amdgcn_wave_barrier();

    float dM[8], XM[8], mM[8];
#pragma unroll
    for (int v = 0; v < 8; ++v) {
      int M = v + hiH * 8;
      dM[v] = scD[M]; XM[v] = scX[M]; mM[v] = scM[M];
    }

    // ---- decay h = gh*h, stage f16 copy for A fragments ----
#pragma unroll
    for (int ct = 0; ct < 4; ++ct)
#pragma unroll
      for (int v = 0; v < 8; ++v) {
        int M = v + hiH * 8;
        float gh = __expf(-fmaxf(dM[v] * ghw[ct] + ghb[ct], 0.f));
        float h = hC[ct][v] * gh;
        hC[ct][v] = h;
        hA[M * HH + ct * 16 + jl] = (_Float16)h;
      }
    __builtin_amdgcn_wave_barrier();
    v16h a0 = ld_afrag_lds(hA, HH, jl, hiH, 0);
    v16h a1 = ld_afrag_lds(hA, HH, jl, hiH, 32);

    // ---- h @ W_hc.T  (z part: nt 0..3, r part: nt 4..7) ----
    v8f zt[4], rt[4];
#pragma unroll
    for (int nt = 0; nt < 4; ++nt) {
      v8f acc = {};
      acc = WMMA_F16(a0, ld_bfrag_lds(sWhc, HH, nt * 16 + jl, hiH * 16), acc);
      acc = WMMA_F16(a1, ld_bfrag_lds(sWhc, HH, nt * 16 + jl, 32 + hiH * 16), acc);
      zt[nt] = acc;
    }
#pragma unroll
    for (int nt = 0; nt < 4; ++nt) {
      v8f acc = {};
      acc = WMMA_F16(a0, ld_bfrag_lds(sWhc, HH, (4 + nt) * 16 + jl, hiH * 16), acc);
      acc = WMMA_F16(a1, ld_bfrag_lds(sWhc, HH, (4 + nt) * 16 + jl, 32 + hiH * 16), acc);
      rt[nt] = acc;
    }

    // ---- gates z, r; stage r*h for second matmul ----
#pragma unroll
    for (int ct = 0; ct < 4; ++ct)
#pragma unroll
      for (int v = 0; v < 8; ++v) {
        int M = v + hiH * 8;
        float z = sigmoidf_(XM[v] * xw0[ct] + xb0[ct] + zt[ct][v] + mM[v] * mw0[ct]);
        float r = sigmoidf_(XM[v] * xw1[ct] + xb1[ct] + rt[ct][v] + mM[v] * mw1[ct]);
        zt[ct][v] = z;
        rA[M * HH + ct * 16 + jl] = (_Float16)(r * hC[ct][v]);
      }
    __builtin_amdgcn_wave_barrier();
    v16h ra0 = ld_afrag_lds(rA, HH, jl, hiH, 0);
    v16h ra1 = ld_afrag_lds(rA, HH, jl, hiH, 32);

    // ---- (r*h) @ W_hn.T, candidate, state update, outputs, scores ----
    float pT[8], pM[8];
#pragma unroll
    for (int v = 0; v < 8; ++v) { pT[v] = 0.f; pM[v] = 0.f; }

#pragma unroll
    for (int ct = 0; ct < 4; ++ct) {
      v8f acc = {};
      acc = WMMA_F16(ra0, ld_bfrag_lds(sWhn, HH, ct * 16 + jl, hiH * 16), acc);
      acc = WMMA_F16(ra1, ld_bfrag_lds(sWhn, HH, ct * 16 + jl, 32 + hiH * 16), acc);
#pragma unroll
      for (int v = 0; v < 8; ++v) {
        int M = v + hiH * 8;
        float htil = tanhf(XM[v] * xw2[ct] + xb2[ct] + acc[v] + mM[v] * mw2[ct]);
        float z = zt[ct][v];
        float hn = (1.f - z) * hC[ct][v] + z * htil;
        hC[ct][v] = hn;
        out_g[(size_t)(rowBase + M) * TT * HH + (size_t)t * HH + ct * 16 + jl] = hn;
        pT[v] += hn * stw[ct];
        pM[v] += hn * smw[ct];
      }
    }

    // reduce attention scores across the 16-lane half (all lanes get row sums)
#pragma unroll
    for (int v = 0; v < 8; ++v) {
#pragma unroll
      for (int msk = 1; msk < 16; msk <<= 1) {
        pT[v] += __shfl_xor(pT[v], msk, 32);
        pM[v] += __shfl_xor(pM[v], msk, 32);
      }
      float eT = __expf(pT[v]);
      float eM = __expf(pM[v]);
      ZT[v] += eT; ZM[v] += eM;
      pT[v] = eT;  pM[v] = eM;
    }
#pragma unroll
    for (int ct = 0; ct < 4; ++ct)
#pragma unroll
      for (int v = 0; v < 8; ++v) {
        accT[ct][v] += pT[v] * hC[ct][v];
        accM[ct][v] += pM[v] * hC[ct][v];
      }
  }

  // pooled hidden states
#pragma unroll
  for (int ct = 0; ct < 4; ++ct)
#pragma unroll
    for (int v = 0; v < 8; ++v) {
      int M = v + hiH * 8;
      hidT_g[(size_t)(rowBase + M) * HH + ct * 16 + jl] = accT[ct][v] / ZT[v];
      hidM_g[(size_t)(rowBase + M) * HH + ct * 16 + jl] = accM[ct][v] / ZM[v];
    }
}

// ---------------------------------------------------------------------------
// Phase 2: MLP 64 -> 300 -> 300 -> 1 with sigmoid range mapping. 16 rows/wave.
// ---------------------------------------------------------------------------
__global__ __launch_bounds__(128) void mlp_kernel(
    const float* __restrict__ hid,                             // [B,64]
    const float* __restrict__ W1, const float* __restrict__ b1, // (300,64),(300)
    const float* __restrict__ W2, const float* __restrict__ b2, // (300,300),(300)
    const float* __restrict__ W3, const float* __restrict__ b3, // (1,300),(1)
    float* __restrict__ dst, float lo, float span)
{
  __shared__ _Float16 sH1[4][16 * KP2];

  const int lane = threadIdx.x & 31;
  const int wv   = threadIdx.x >> 5;
  const int jl   = lane & 15;
  const int hiH  = lane >> 4;
  const int rowBase = (blockIdx.x * 4 + wv) * 16;
  _Float16* h1 = sH1[wv];

  // zero the staging buffer (covers K padding 300..319)
  for (int i = lane; i < 16 * KP2 / 8; i += 32) {
    h8 z = {};
    *(h8*)(h1 + i * 8) = z;
  }
  __builtin_amdgcn_wave_barrier();

  // layer-1 A fragments straight from global hid rows
  const float* hrow = hid + (size_t)(rowBase + jl) * HH;
  v16h a0 = ld_afrag_gf32(hrow, hiH, 0);
  v16h a1 = ld_afrag_gf32(hrow, hiH, 32);

  // ---- layer 1: 19 output tiles of 16 (300 padded to 304) ----
#pragma unroll 1
  for (int nt = 0; nt < 19; ++nt) {
    int n = nt * 16 + jl;
    bool valid = (n < REGD);
    v16h bw0 = zero_v16h(), bw1 = zero_v16h();
    if (valid) {
      bw0 = ld_bfrag_g(W1, HH, n, hiH * 16);
      bw1 = ld_bfrag_g(W1, HH, n, 32 + hiH * 16);
    }
    v8f acc = {};
    acc = WMMA_F16(a0, bw0, acc);
    acc = WMMA_F16(a1, bw1, acc);
    float bias = valid ? b1[n] : 0.f;
#pragma unroll
    for (int v = 0; v < 8; ++v) {
      int M = v + hiH * 8;
      float y = valid ? eluf_(acc[v] + bias) : 0.f;
      h1[M * KP2 + nt * 16 + jl] = (_Float16)y;
    }
  }
  __builtin_amdgcn_wave_barrier();

  // ---- layer 2 (K=300 padded to 320) fused with layer-3 dot ----
  float dacc[8];
#pragma unroll
  for (int v = 0; v < 8; ++v) dacc[v] = 0.f;

#pragma unroll 1
  for (int nt2 = 0; nt2 < 19; ++nt2) {
    int n2 = nt2 * 16 + jl;
    bool nv = (n2 < REGD);
    v8f acc = {};
#pragma unroll 1
    for (int kc = 0; kc < 10; ++kc) {
      v16h a = ld_afrag_lds(h1, KP2, jl, hiH, kc * 32);
      v16h bw = zero_v16h();
      if (nv) {
        if (kc < 9) {
          bw = ld_bfrag_g(W2, REGD, n2, kc * 32 + hiH * 16);
        } else {
          int kb = kc * 32 + hiH * 16;
#pragma unroll
          for (int i = 0; i < 16; ++i) {
            int k = kb + i;
            bw[i] = (_Float16)((k < REGD) ? W2[(size_t)n2 * REGD + k] : 0.f);
          }
        }
      }
      acc = WMMA_F16(a, bw, acc);
    }
    float bias = nv ? b2[n2] : 0.f;
    float w3   = nv ? W3[n2] : 0.f;
#pragma unroll
    for (int v = 0; v < 8; ++v) {
      float y = eluf_(acc[v] + bias);
      dacc[v] += y * w3;
    }
  }

  // reduce layer-3 dot across 16-lane half, apply sigmoid range, store
  float bb3 = b3[0];
#pragma unroll
  for (int v = 0; v < 8; ++v) {
#pragma unroll
    for (int msk = 1; msk < 16; msk <<= 1) dacc[v] += __shfl_xor(dacc[v], msk, 32);
    float res = lo + span * sigmoidf_(dacc[v] + bb3);
    if (jl == 0) dst[rowBase + v + hiH * 8] = res;
  }
}

// ---------------------------------------------------------------------------
// Phase 3: closed-form signal equation (elementwise over B*T)
// ---------------------------------------------------------------------------
__global__ void finalize_kernel(const float* __restrict__ fa, const float* __restrict__ TR,
                                const float* __restrict__ T10, const float* __restrict__ M0,
                                float* __restrict__ Xout)
{
  int idx = blockIdx.x * blockDim.x + threadIdx.x;
  if (idx >= BATCH * TT) return;
  int b = idx / TT;
  float t10 = T10[b];
  float m0  = M0[b];
  float e = __expf(-TR[b] / t10);
  float f = fa[idx];
  float s = __sinf(f);
  float c = __cosf(f);
  Xout[idx] = (1.f - e) * s / (1.f - c * e) * m0;
}

extern "C" void kernel_launch(void* const* d_in, const int* in_sizes, int n_in,
                              void* d_out, int out_size, void* d_ws, size_t ws_size,
                              hipStream_t stream) {
  const float* Xfa    = (const float*)d_in[0];
  const float* Xlast  = (const float*)d_in[1];
  const float* Xmean  = (const float*)d_in[2];
  const float* Xmask  = (const float*)d_in[3];
  const float* Xdelta = (const float*)d_in[4];
  const float* fa     = (const float*)d_in[5];
  const float* TR     = (const float*)d_in[6];
  const float* Wgx = (const float*)d_in[7];
  const float* bgx = (const float*)d_in[8];
  const float* Wgh = (const float*)d_in[9];
  const float* bgh = (const float*)d_in[10];
  const float* Wxc = (const float*)d_in[11];
  const float* bxc = (const float*)d_in[12];
  const float* Whn = (const float*)d_in[13];
  const float* Whc = (const float*)d_in[14];
  const float* Wmc = (const float*)d_in[15];
  const float* WsT = (const float*)d_in[16];
  const float* WsM = (const float*)d_in[17];
  const float* Wt1 = (const float*)d_in[18];
  const float* bt1 = (const float*)d_in[19];
  const float* Wt2 = (const float*)d_in[20];
  const float* bt2 = (const float*)d_in[21];
  const float* Wt3 = (const float*)d_in[22];
  const float* bt3 = (const float*)d_in[23];
  const float* Wm1 = (const float*)d_in[24];
  const float* bm1 = (const float*)d_in[25];
  const float* Wm2 = (const float*)d_in[26];
  const float* bm2 = (const float*)d_in[27];
  const float* Wm3 = (const float*)d_in[28];
  const float* bm3 = (const float*)d_in[29];

  float* out = (float*)d_out;
  const size_t XOUT_OFF = 0;
  const size_t T10_OFF  = (size_t)BATCH * TT;
  const size_t M0_OFF   = T10_OFF + BATCH;
  const size_t OUT_OFF  = M0_OFF + BATCH;
  const size_t XGRU_OFF = OUT_OFF + (size_t)BATCH * TT * HH;

  float* hidT = (float*)d_ws;                     // [B,64]
  float* hidM = hidT + (size_t)BATCH * HH;        // [B,64]

  grud_scan_kernel<<<BATCH / (4 * 16), 128, 0, stream>>>(
      Xfa, Xlast, Xmean, Xmask, Xdelta,
      Wgx, bgx, Wgh, bgh, Wxc, bxc, Whn, Whc, Wmc, WsT, WsM,
      out + OUT_OFF, out + XGRU_OFF, hidT, hidM);

  mlp_kernel<<<BATCH / 64, 128, 0, stream>>>(hidT, Wt1, bt1, Wt2, bt2, Wt3, bt3,
                                             out + T10_OFF, 0.2f, 4.8f);
  mlp_kernel<<<BATCH / 64, 128, 0, stream>>>(hidM, Wm1, bm1, Wm2, bm2, Wm3, bm3,
                                             out + M0_OFF, 0.1f, 19.9f);

  finalize_kernel<<<(BATCH * TT + 255) / 256, 256, 0, stream>>>(
      fa, TR, out + T10_OFF, out + M0_OFF, out + XOUT_OFF);
}
